// CrossAttention_84610855731259
// MI455X (gfx1250) — compile-verified, matrix operand order checked by reading, not run on hardware
//
#include <hip/hip_runtime.h>
#include <math.h>

typedef __attribute__((ext_vector_type(16))) __bf16 v16bf;
typedef __attribute__((ext_vector_type(8)))  float  v8f;
typedef int v4i_gcc __attribute__((vector_size(4 * sizeof(int))));  // matches builtin param

#define BATCH 8
#define C     96
#define HH    192
#define WW    192
#define NPIX  (HH*WW)      // 36864
#define HEADS 3
#define HC    32
#define KSPLIT 64
#define NCHUNK (NPIX / KSPLIT)   // 576
#define LDA   100                 // padded LDS row stride for A (bank de-conflict)

// ---- async global->LDS staging (gfx1250), guarded so either toolchain compiles
#if defined(__has_builtin)
#  if __has_builtin(__builtin_amdgcn_global_load_async_to_lds_b128) && \
      __has_builtin(__builtin_amdgcn_s_wait_asynccnt)
#    define USE_ASYNC_LDS 1
#  endif
#endif
#ifndef USE_ASYNC_LDS
#  define USE_ASYNC_LDS 0
#endif

#define AS1 __attribute__((address_space(1)))
#define AS3 __attribute__((address_space(3)))

static __device__ __forceinline__ AS3 v4i_gcc* lds_cast4(void* p) {
    return (AS3 v4i_gcc*)(unsigned int)(unsigned long long)p;  // LDS offset lives in low 32 bits
}
static __device__ __forceinline__ AS1 v4i_gcc* glob_cast4(const void* p) {
    return (AS1 v4i_gcc*)(unsigned long long)p;
}

// ---------------------------------------------------------------------------
// Zero-init a float region (atomic accumulators live here).
__global__ void zero_kernel(float* __restrict__ p, long n) {
    long i = (long)blockIdx.x * blockDim.x + threadIdx.x;
    if (i < n) p[i] = 0.0f;
}

// ---------------------------------------------------------------------------
// O[b, m, n] = sum_k W[(b), m, k] * X[b, k, n]   (M=96, K=96, N=36864)
// One wave owns a full 96-row x 16-col strip: 6 WMMA C-tiles, B fragment loaded
// once per K-step and reused 6x -> X is read exactly once from HBM.
// Weights are staged once per block into (padded) LDS, async when available.
__global__ __launch_bounds__(256)
void gemm_pw_wmma(const float* __restrict__ W, long wBatchStride,
                  const float* __restrict__ X, float* __restrict__ O)
{
    __shared__ float lA[C * LDA];                 // 96 x 100 floats = 37.5 KB

    const int tid  = threadIdx.x;
    const int lane = tid & 31;
    const int wave = tid >> 5;
    const int b    = blockIdx.y;

    const float* Wb = W + (long)b * wBatchStride;
    const float* Xb = X + (long)b * C * NPIX;
    float*       Ob = O + (long)b * C * NPIX;

    // ---- stage A (96x96) into LDS with padded rows
#if USE_ASYNC_LDS
    for (int i = tid * 4; i < C * C; i += 256 * 4) {
        const int m = i / C, k = i % C;           // 4-float chunks never cross rows (96 % 4 == 0)
        __builtin_amdgcn_global_load_async_to_lds_b128(
            glob_cast4(Wb + i), lds_cast4(&lA[m * LDA + k]), 0, 0);
    }
    __builtin_amdgcn_s_wait_asynccnt(0);
#else
    for (int i = tid; i < C * C; i += 256) {
        const int m = i / C, k = i % C;
        lA[m * LDA + k] = Wb[i];
    }
#endif
    __syncthreads();

    const long n0  = (long)(blockIdx.x * 8 + wave) * 16;
    const int col  = lane & 15;                   // N (B,C,D) / M (A) index within tile
    const int kb   = (lane >> 4) * 8;             // K sub-offset for this half-wave

    v8f acc[6] = { {}, {}, {}, {}, {}, {} };

    for (int kt = 0; kt < C; kt += 32) {
        // B fragment: X[kt+kb+e, n0+col] (e<8) and X[kt+kb+16+e, n0+col]
        const float* xp = Xb + (long)(kt + kb) * NPIX + n0 + col;
        v16bf bm;
        #pragma unroll
        for (int e = 0; e < 8; ++e) {
            bm[e]     = (__bf16)xp[(long)e * NPIX];
            bm[8 + e] = (__bf16)xp[(long)(16 + e) * NPIX];
        }
        #pragma unroll
        for (int mt = 0; mt < 6; ++mt) {
            const float* ap = &lA[(mt * 16 + col) * LDA + kt + kb];
            v16bf a;
            #pragma unroll
            for (int e = 0; e < 8; ++e) { a[e] = (__bf16)ap[e]; a[8 + e] = (__bf16)ap[16 + e]; }
            acc[mt] = __builtin_amdgcn_wmma_f32_16x16x32_bf16(
                false, a, false, bm, (short)0, acc[mt], false, false);
        }
    }

    const int mhalf = (lane >> 4) * 8;
    #pragma unroll
    for (int mt = 0; mt < 6; ++mt) {
        #pragma unroll
        for (int r = 0; r < 8; ++r)
            Ob[(long)(mt * 16 + mhalf + r) * NPIX + n0 + col] = acc[mt][r];
    }
}

// ---------------------------------------------------------------------------
// Out = Xres + depthwise3x3(Hbuf) (pad 1); optionally accumulate sum(out^2)
// per (b,c) row into normsq (block reduce + 1 atomic per block).
__global__ __launch_bounds__(256)
void dw_res_kernel(const float* __restrict__ Hbuf, const float* __restrict__ Xres,
                   const float* __restrict__ DW, float* __restrict__ Out,
                   float* __restrict__ normsq)
{
    __shared__ float red[256];
    const int pix = blockIdx.x * 256 + threadIdx.x;   // < NPIX
    const int c   = blockIdx.y;
    const int b   = blockIdx.z;
    const int hy  = pix / WW;
    const int wx  = pix % WW;
    const long plane = ((long)b * C + c) * NPIX;
    const float* hp = Hbuf + plane;
    const float* kw = DW + c * 9;

    float acc = Xres[plane + pix];
    #pragma unroll
    for (int dy = -1; dy <= 1; ++dy) {
        const int yy = hy + dy;
        if (yy < 0 || yy >= HH) continue;
        #pragma unroll
        for (int dx = -1; dx <= 1; ++dx) {
            const int xx = wx + dx;
            if (xx < 0 || xx >= WW) continue;
            acc += kw[(dy + 1) * 3 + (dx + 1)] * hp[yy * WW + xx];
        }
    }
    Out[plane + pix] = acc;

    if (normsq) {
        red[threadIdx.x] = acc * acc;
        __syncthreads();
        for (int s = 128; s > 0; s >>= 1) {
            if (threadIdx.x < s) red[threadIdx.x] += red[threadIdx.x + s];
            __syncthreads();
        }
        if (threadIdx.x == 0) atomicAdd(&normsq[b * C + c], red[0]);
    }
}

// ---------------------------------------------------------------------------
// G[b,head][c,d] += sum_n q[b,head*32+c,n] * k[b,head*32+d,n]  (split-K WMMA)
// Block = 4 waves, each owning one 16x16 tile of the 32x32 Gram.
__global__ __launch_bounds__(128)
void gram_wmma(const float* __restrict__ Q, const float* __restrict__ K,
               float* __restrict__ G)
{
    const int lane = threadIdx.x & 31;
    const int wave = threadIdx.x >> 5;         // 0..3
    const int bh   = blockIdx.x;               // b*HEADS + head
    const int b    = bh / HEADS;
    const int head = bh % HEADS;
    const int ks   = blockIdx.y;               // 0..KSPLIT-1
    const int ct   = (wave >> 1) * 16;
    const int dt   = (wave & 1) * 16;

    const float* qb  = Q + ((long)b * C + head * HC) * NPIX;
    const float* kbp = K + ((long)b * C + head * HC) * NPIX;

    const int row = lane & 15;
    const int kb  = (lane >> 4) * 8;
    const long n0 = (long)ks * NCHUNK;

    v8f acc = {};
    for (int nc = 0; nc < NCHUNK; nc += 32) {
        const long nbase = n0 + nc + kb;
        const float* qp = qb  + (long)(ct + row) * NPIX + nbase;
        const float* kp = kbp + (long)(dt + row) * NPIX + nbase;
        v16bf a, bm;
        #pragma unroll
        for (int e = 0; e < 8; ++e) {
            a[e]      = (__bf16)qp[e];      a[8 + e]  = (__bf16)qp[16 + e];
            bm[e]     = (__bf16)kp[e];      bm[8 + e] = (__bf16)kp[16 + e];
        }
        acc = __builtin_amdgcn_wmma_f32_16x16x32_bf16(
            false, a, false, bm, (short)0, acc, false, false);
    }

    float* g = G + (long)bh * HC * HC;
    const int dcol  = dt + (lane & 15);
    const int rbase = ct + (lane >> 4) * 8;
    #pragma unroll
    for (int r = 0; r < 8; ++r)
        atomicAdd(&g[(rbase + r) * HC + dcol], acc[r]);
}

// ---------------------------------------------------------------------------
// attn[b,head][c,:] = softmax( G[c,d] / (max(|q_c|,eps)*max(|k_d|,eps)) * T[head] )
__global__ void softmax_kernel(const float* __restrict__ G,
                               const float* __restrict__ nsq_q,
                               const float* __restrict__ nsq_k,
                               const float* __restrict__ temp,
                               float* __restrict__ A)
{
    const int bh = blockIdx.x;
    const int b = bh / HEADS, head = bh % HEADS;
    const int crow = threadIdx.x;              // 0..31
    const float* g = G + (long)bh * HC * HC;
    float*       a = A + (long)bh * HC * HC;
    const float t = temp[head];
    const float nq = fmaxf(sqrtf(nsq_q[b * C + head * HC + crow]), 1e-12f);

    float vals[HC];
    float mx = -3.4e38f;
    for (int d = 0; d < HC; ++d) {
        const float nk = fmaxf(sqrtf(nsq_k[b * C + head * HC + d]), 1e-12f);
        const float v = g[crow * HC + d] / (nq * nk) * t;
        vals[d] = v;
        mx = fmaxf(mx, v);
    }
    float s = 0.0f;
    for (int d = 0; d < HC; ++d) { vals[d] = expf(vals[d] - mx); s += vals[d]; }
    const float inv = 1.0f / s;
    for (int d = 0; d < HC; ++d) a[crow * HC + d] = vals[d] * inv;
}

// ---------------------------------------------------------------------------
// M_b[co, head*32+d] = sum_c proj[co, head*32+c] * attn[b,head][c,d]
__global__ void mbuild_kernel(const float* __restrict__ proj,
                              const float* __restrict__ A,
                              float* __restrict__ M)
{
    const int idx = blockIdx.x * 256 + threadIdx.x;
    if (idx >= BATCH * C * C) return;
    const int b    = idx / (C * C);
    const int rem  = idx % (C * C);
    const int co   = rem / C;
    const int cpp  = rem % C;
    const int head = cpp / HC, d = cpp % HC;
    const float* a = A + ((long)b * HEADS + head) * HC * HC;
    const float* p = proj + co * C + head * HC;
    float s = 0.0f;
    for (int cc = 0; cc < HC; ++cc) s += p[cc] * a[cc * HC + d];
    M[idx] = s;
}

// ---------------------------------------------------------------------------
extern "C" void kernel_launch(void* const* d_in, const int* in_sizes, int n_in,
                              void* d_out, int out_size, void* d_ws, size_t ws_size,
                              hipStream_t stream)
{
    const float* x    = (const float*)d_in[0];
    const float* y    = (const float*)d_in[1];
    const float* z    = (const float*)d_in[2];
    const float* temp = (const float*)d_in[3];
    const float* q_pw = (const float*)d_in[4];
    const float* q_dw = (const float*)d_in[5];
    const float* k_pw = (const float*)d_in[6];
    const float* k_dw = (const float*)d_in[7];
    const float* v_pw = (const float*)d_in[8];
    const float* v_dw = (const float*)d_in[9];
    const float* proj = (const float*)d_in[10];
    float* out = (float*)d_out;

    const long PLANE = (long)BATCH * C * NPIX;     // 28,311,552 floats
    float* ws    = (float*)d_ws;
    float* bufH  = ws;                              // pointwise conv result
    float* bufQ  = ws + PLANE;                      // q, later reused for v
    float* bufK  = ws + 2 * PLANE;                  // k
    float* G     = ws + 3 * PLANE;                  // 24*1024
    float* nsq_q = G + (long)BATCH * HEADS * HC * HC;   // 768
    float* nsq_k = nsq_q + BATCH * C;                   // 768
    float* attn  = nsq_k + BATCH * C;                   // 24*1024
    float* Mb    = attn + (long)BATCH * HEADS * HC * HC; // 8*96*96

    const dim3 gGrid(NPIX / 16 / 8, BATCH);         // (288, 8): 8 waves x 16-col strips
    const dim3 gBlk(256);
    const dim3 dGrid(NPIX / 256, C, BATCH);         // (144, 96, 8)
    const dim3 dBlk(256);

    // zero atomic accumulators: G | nsq_q | nsq_k (contiguous)
    const long zn = (long)BATCH * HEADS * HC * HC + 2L * BATCH * C;
    zero_kernel<<<(int)((zn + 255) / 256), 256, 0, stream>>>(G, zn);

    // q = x + dw(pw(x)); fused rowwise sum(q^2)
    gemm_pw_wmma<<<gGrid, gBlk, 0, stream>>>(q_pw, 0, x, bufH);
    dw_res_kernel<<<dGrid, dBlk, 0, stream>>>(bufH, x, q_dw, bufQ, nsq_q);
    // k = y + dw(pw(y)); fused rowwise sum(k^2)
    gemm_pw_wmma<<<gGrid, gBlk, 0, stream>>>(k_pw, 0, y, bufH);
    dw_res_kernel<<<dGrid, dBlk, 0, stream>>>(bufH, y, k_dw, bufK, nsq_k);
    // Gram (split-K WMMA, atomic accumulate)
    gram_wmma<<<dim3(BATCH * HEADS, KSPLIT), 128, 0, stream>>>(bufQ, bufK, G);
    // normalized, temperature-scaled softmax -> attn
    softmax_kernel<<<BATCH * HEADS, HC, 0, stream>>>(G, nsq_q, nsq_k, temp, attn);
    // v = z + dw(pw(z)) (reuses bufQ; q no longer needed)
    gemm_pw_wmma<<<gGrid, gBlk, 0, stream>>>(v_pw, 0, z, bufH);
    dw_res_kernel<<<dGrid, dBlk, 0, stream>>>(bufH, z, v_dw, bufQ, nullptr);
    // fold proj @ blockdiag(attn) into per-batch 96x96 weight
    mbuild_kernel<<<(BATCH * C * C + 255) / 256, 256, 0, stream>>>(proj, attn, Mb);
    // out = M_b @ v (single WMMA GEMM pass, writes d_out)
    gemm_pw_wmma<<<gGrid, gBlk, 0, stream>>>(Mb, (long)C * C, bufQ, out);
}
